// TreeAttnBatch_84061099917529
// MI455X (gfx1250) — matrix operand order repaired and checked by reading, not compiled, not run on hardware
//
#include <hip/hip_runtime.h>
#include <hip/hip_bf16.h>

// ---------------------------------------------------------------------------
// Types for CDNA5 WMMA (wave32): v_wmma_f32_16x16x32_bf16
// ---------------------------------------------------------------------------
typedef __attribute__((ext_vector_type(16))) __bf16 v16bf;
typedef __attribute__((ext_vector_type(8)))  __bf16 v8bf;
typedef __attribute__((ext_vector_type(4)))  __bf16 v4bf;
typedef __attribute__((ext_vector_type(8)))  float  v8f;
typedef __attribute__((ext_vector_type(4)))  float  v4f;

#define BATCH   1024
#define NODES   63
#define HLX     256
#define HMX     128
#define XDIM    640   // 5*HM

__device__ __constant__ int d_OFFS[6] = {0, 32, 48, 56, 60, 62};

__device__ __forceinline__ __bf16 f2bf(float f) {
  unsigned u = __builtin_bit_cast(unsigned, f);
  unsigned r = u + 0x7FFFu + ((u >> 16) & 1u);   // round-to-nearest-even
  unsigned short hs = (unsigned short)(r >> 16);
  return __builtin_bit_cast(__bf16, hs);
}

__device__ __forceinline__ v4bf f2bf4(v4f v) {
  v4bf o;
  o.x = f2bf(v.x); o.y = f2bf(v.y); o.z = f2bf(v.z); o.w = f2bf(v.w);
  return o;
}

__device__ __forceinline__ float sigmoidf_(float x) {
  return 1.0f / (1.0f + __expf(-x));
}

// ---------------------------------------------------------------------------
// Generic bf16 WMMA GEMM:  C[M,N] (fp32) = act( (A@B [+ bias] [+ C]) [* rowmask] )
//   REQUIRES: M % 64 == 0, N % 64 == 0, lda % 4 == 0, (ldb % 4 == 0 if transB).
//   K may be arbitrary (zero-padded in the final K-step).
//   A fp32; if ashift >= 0, row(m) = (m>>ashift)*63 + aoff + (m & ((1<<ashift)-1))
//   B fp32; transB=1 means logical B[k][n] = Bm[n*ldb + k]
// Tile: 64(M) x 64(N), K step 32, 128 threads = 4 waves, each wave 16x64.
// ---------------------------------------------------------------------------
__global__ __launch_bounds__(128)
void gemm_bf16_kernel(const float* __restrict__ A, int lda, int ashift, int aoff,
                      const float* __restrict__ Bm, int ldb, int transB,
                      float* __restrict__ C, int ldc,
                      const float* __restrict__ bias,
                      const float* __restrict__ rowmask,
                      int M, int N, int K, int accum, int act)
{
  __shared__ __align__(64) __bf16 As[64 * 32];
  __shared__ __align__(64) __bf16 Bs[64 * 32];

  const int tid  = threadIdx.x;
  const int wave = tid >> 5;
  const int lane = tid & 31;
  const int bm   = blockIdx.y * 64;
  const int bn   = blockIdx.x * 64;
  const int amask = (ashift >= 0) ? ((1 << ashift) - 1) : 0;

  // Per-thread staging coordinates: 512 float4 groups, 4 per thread.
  const int rA  = (tid) >> 3;          // base row for i=0 (rows advance by 16/i)
  const int kqA = (tid & 7) * 4;

  // Precompute the (possibly remapped) A row pointers for this thread's 4 rows.
  const float* aptr[4];
#pragma unroll
  for (int i = 0; i < 4; ++i) {
    int r  = rA + i * 16;
    int gm = bm + r;
    int ar = (ashift >= 0) ? (((gm >> ashift) * NODES) + aoff + (gm & amask)) : gm;
    aptr[i] = A + (long)ar * lda + kqA;
  }

  v8f acc[4];
#pragma unroll
  for (int t = 0; t < 4; ++t)
    acc[t] = (v8f){0.f, 0.f, 0.f, 0.f, 0.f, 0.f, 0.f, 0.f};

  for (int k0 = 0; k0 < K; k0 += 32) {
    const bool full = (k0 + 32 <= K);   // uniform; false only on the K tail
    if (full) {
      // ---- fast path: unguarded float4 loads ----
#pragma unroll
      for (int i = 0; i < 4; ++i) {
        v4f av = *(const v4f*)(aptr[i] + k0);
        if (k0 + 32 < K) __builtin_prefetch(aptr[i] + k0 + 32, 0, 0);
        *(v4bf*)(As + (rA + i * 16) * 32 + kqA) = f2bf4(av);
      }
      if (transB) {
#pragma unroll
        for (int i = 0; i < 4; ++i) {
          int cn = rA + i * 16;
          v4f bv = *(const v4f*)(Bm + (long)(bn + cn) * ldb + k0 + kqA);
          *(v4bf*)(Bs + cn * 32 + kqA) = f2bf4(bv);
        }
      } else {
#pragma unroll
        for (int i = 0; i < 16; ++i) {
          int e  = tid + i * 128;
          int cn = e & 63;
          int k  = e >> 6;
          Bs[cn * 32 + k] = f2bf(Bm[(long)(k0 + k) * ldb + bn + cn]);
        }
      }
    } else {
      // ---- tail path (once per kernel at most): element-guarded ----
#pragma unroll
      for (int i = 0; i < 4; ++i) {
        v4f av = (v4f){0.f, 0.f, 0.f, 0.f};
        const float* ap = aptr[i] + k0;
        int gk = k0 + kqA;
        if (gk + 0 < K) av.x = ap[0];
        if (gk + 1 < K) av.y = ap[1];
        if (gk + 2 < K) av.z = ap[2];
        if (gk + 3 < K) av.w = ap[3];
        *(v4bf*)(As + (rA + i * 16) * 32 + kqA) = f2bf4(av);
      }
      if (transB) {
#pragma unroll
        for (int i = 0; i < 4; ++i) {
          int cn = rA + i * 16;
          const float* bp = Bm + (long)(bn + cn) * ldb + k0 + kqA;
          int gk = k0 + kqA;
          v4f bv = (v4f){0.f, 0.f, 0.f, 0.f};
          if (gk + 0 < K) bv.x = bp[0];
          if (gk + 1 < K) bv.y = bp[1];
          if (gk + 2 < K) bv.z = bp[2];
          if (gk + 3 < K) bv.w = bp[3];
          *(v4bf*)(Bs + cn * 32 + kqA) = f2bf4(bv);
        }
      } else {
#pragma unroll
        for (int i = 0; i < 16; ++i) {
          int e  = tid + i * 128;
          int cn = e & 63;
          int k  = e >> 6;
          float v = (k0 + k < K) ? Bm[(long)(k0 + k) * ldb + bn + cn] : 0.f;
          Bs[cn * 32 + k] = f2bf(v);
        }
      }
    }
    __syncthreads();

    // A fragment: lane<16 -> row=lane, K 0..15; lane>=16 -> K 16..31
    const int row = wave * 16 + (lane & 15);
    const int ka  = (lane < 16) ? 0 : 16;
    v8bf alo = *(const v8bf*)(As + row * 32 + ka);
    v8bf ahi = *(const v8bf*)(As + row * 32 + 8 + ka);
    v16bf af = __builtin_shufflevector(alo, ahi,
        0, 1, 2, 3, 4, 5, 6, 7, 8, 9, 10, 11, 12, 13, 14, 15);

    const int kb = (lane < 16) ? 0 : 16;   // B: lane<16 holds K 0..15 of its column
    v16bf bfr[4];
#pragma unroll
    for (int t = 0; t < 4; ++t)
      bfr[t] = *(const v16bf*)(Bs + (t * 16 + (lane & 15)) * 32 + kb);
#pragma unroll
    for (int t = 0; t < 4; ++t)
      acc[t] = __builtin_amdgcn_wmma_f32_16x16x32_bf16(
          false, af, false, bfr[t], (short)0, acc[t], false, false);
    __syncthreads();
  }

  // Epilogue (no M/N guards: both are multiples of 64 by contract).
  const int rbase = bm + wave * 16 + ((lane < 16) ? 0 : 8);
  const int cb    = bn + (lane & 15);
#pragma unroll
  for (int t = 0; t < 4; ++t) {
    int gn = cb + t * 16;
#pragma unroll
    for (int r = 0; r < 8; ++r) {
      int gm = rbase + r;
      float v = acc[t][r];
      if (bias)    v += bias[gn];
      if (accum)   v += C[(long)gm * ldc + gn];
      if (rowmask) v *= rowmask[gm];
      if (act == 1)      v = tanhf(v);
      else if (act == 2) v = fmaxf(v, 0.f);
      C[(long)gm * ldc + gn] = v;
    }
  }
}

// ---------------------------------------------------------------------------
// Attention: one block per (b, node j) of level l.
// e = tanh(desc @ W_wh + ext), logits = e @ W_a, softmax over m,
// pooled = sum_t w[t] * desc[t].    desc rows padded to 64 with zeros
// (zeroing done arithmetically: clamped index + select multiply).
// ---------------------------------------------------------------------------
__global__ __launch_bounds__(128)
void attention_kernel(const float* __restrict__ hbuf,
                      const float* __restrict__ extbuf,
                      const float* __restrict__ W_wh,
                      const float* __restrict__ W_a,
                      float* __restrict__ pooled,
                      int l, int nshift, int m)
{
  __shared__ __align__(64) __bf16 As[64 * 32];
  __shared__ __align__(64) __bf16 Bs[64 * 32];
  __shared__ float red[64 * 16];
  __shared__ float logits[64];
  __shared__ float wgt[64];
  __shared__ float wa[64];
  __shared__ int   nodeIdx[64];

  const int bj   = blockIdx.x;             // b*n + j
  const int b    = bj >> nshift;
  const int j    = bj & ((1 << nshift) - 1);
  const int tid  = threadIdx.x;
  const int wave = tid >> 5;
  const int lane = tid & 31;

  if (tid < 64) {
    wa[tid] = W_a[tid];
    int nd = -1;
    if (tid < m) {
      int rem = tid;
      for (int lp = 0; lp < l; ++lp) {
        int cnt = 1 << (l - lp);
        if (rem < cnt) { nd = d_OFFS[lp] + j * cnt + rem; break; }
        rem -= cnt;
      }
    }
    nodeIdx[tid] = nd;
  }
  __syncthreads();

  const int rA  = tid >> 3;
  const int kqA = (tid & 7) * 4;
  // Per-thread: 4 desc rows, clamped pointer + zero-select
  const float* aptr[4];
  float asel[4];
#pragma unroll
  for (int i = 0; i < 4; ++i) {
    int nd = nodeIdx[rA + i * 16];
    asel[i] = (nd >= 0) ? 1.f : 0.f;
    int ndc = (nd >= 0) ? nd : 0;
    aptr[i] = hbuf + (((long)b * NODES + ndc) << 8) + kqA;
  }

  v8f acc[4];
#pragma unroll
  for (int t = 0; t < 4; ++t)
    acc[t] = (v8f){0.f, 0.f, 0.f, 0.f, 0.f, 0.f, 0.f, 0.f};

  for (int k0 = 0; k0 < HLX; k0 += 32) {
#pragma unroll
    for (int i = 0; i < 4; ++i) {
      v4f av = *(const v4f*)(aptr[i] + k0);
      av = av * asel[i];
      *(v4bf*)(As + (rA + i * 16) * 32 + kqA) = f2bf4(av);
    }
#pragma unroll
    for (int i = 0; i < 16; ++i) {
      int e  = tid + i * 128;
      int cn = e & 63;
      int k  = e >> 6;
      Bs[cn * 32 + k] = f2bf(W_wh[(long)(k0 + k) * 64 + cn]);
    }
    __syncthreads();

    const int row = wave * 16 + (lane & 15);
    const int ka  = (lane < 16) ? 0 : 16;
    v8bf alo = *(const v8bf*)(As + row * 32 + ka);
    v8bf ahi = *(const v8bf*)(As + row * 32 + 8 + ka);
    v16bf af = __builtin_shufflevector(alo, ahi,
        0, 1, 2, 3, 4, 5, 6, 7, 8, 9, 10, 11, 12, 13, 14, 15);
    const int kb = (lane < 16) ? 0 : 16;
    v16bf bfr[4];
#pragma unroll
    for (int t = 0; t < 4; ++t)
      bfr[t] = *(const v16bf*)(Bs + (t * 16 + (lane & 15)) * 32 + kb);
#pragma unroll
    for (int t = 0; t < 4; ++t)
      acc[t] = __builtin_amdgcn_wmma_f32_16x16x32_bf16(
          false, af, false, bfr[t], (short)0, acc[t], false, false);
    __syncthreads();
  }

  // logits: per-lane partial dot of tanh(e + ext) with W_a over its 4 columns
  const int colb = lane & 15;
  float part[8];
#pragma unroll
  for (int r = 0; r < 8; ++r) part[r] = 0.f;
#pragma unroll
  for (int t = 0; t < 4; ++t) {
    int col = t * 16 + colb;
    float ev  = extbuf[(long)bj * 64 + col];
    float wav = wa[col];
#pragma unroll
    for (int r = 0; r < 8; ++r)
      part[r] += tanhf(acc[t][r] + ev) * wav;
  }
  const int rb = wave * 16 + ((lane < 16) ? 0 : 8);
#pragma unroll
  for (int r = 0; r < 8; ++r) red[(rb + r) * 16 + colb] = part[r];
  __syncthreads();

  if (tid < 64) {
    float s = 0.f;
#pragma unroll
    for (int q = 0; q < 16; ++q) s += red[tid * 16 + q];
    logits[tid] = s;
  }
  __syncthreads();

  if (tid == 0) {
    float mx = -1e30f;
    for (int t = 0; t < m; ++t) mx = fmaxf(mx, logits[t]);
    float s = 0.f;
    for (int t = 0; t < m; ++t) { float e = __expf(logits[t] - mx); wgt[t] = e; s += e; }
    float inv = 1.f / s;
    for (int t = 0; t < m; ++t) wgt[t] *= inv;
  }
  __syncthreads();

  for (int e = tid; e < HLX; e += 128) {
    float a2 = 0.f;
    for (int t = 0; t < m; ++t) {
      int nd = nodeIdx[t];
      a2 += wgt[t] * hbuf[(((long)b * NODES + nd) << 8) + e];
    }
    pooled[(long)bj * HLX + e] = a2;
  }
}

// ---------------------------------------------------------------------------
// c0[m,:] = 0.5 * (c[child 2j] + c[child 2j+1])
// ---------------------------------------------------------------------------
__global__ void c0_kernel(const float* __restrict__ cbuf, float* __restrict__ c0buf,
                          int nshift, int choff, int total)
{
  int idx = blockIdx.x * blockDim.x + threadIdx.x;
  if (idx >= total) return;
  int mrow = idx >> 8, e = idx & 255;
  int b = mrow >> nshift, j = mrow & ((1 << nshift) - 1);
  const float* base = cbuf + (((long)b * NODES + choff + 2 * j) << 8);
  c0buf[idx] = 0.5f * (base[e] + base[HLX + e]);
}

// ---------------------------------------------------------------------------
// LSTM pointwise: gates[m,1024] -> h,c at node off+j.  bhh added when the
// h0@W_hh GEMM was skipped (level 0, h0 == 0).
// ---------------------------------------------------------------------------
__global__ void lstm_pointwise(const float* __restrict__ gates,
                               const float* __restrict__ c0buf, int use_c0,
                               const float* __restrict__ bhh,
                               float* __restrict__ h, float* __restrict__ c,
                               int nshift, int off, int total)
{
  int idx = blockIdx.x * blockDim.x + threadIdx.x;
  if (idx >= total) return;
  int mrow = idx >> 8, e = idx & 255;
  long g = (long)mrow * 1024;
  float i_ = gates[g + e];
  float f_ = gates[g + 256 + e];
  float g_ = gates[g + 512 + e];
  float o_ = gates[g + 768 + e];
  if (bhh) { i_ += bhh[e]; f_ += bhh[256 + e]; g_ += bhh[512 + e]; o_ += bhh[768 + e]; }
  float c0 = use_c0 ? c0buf[idx] : 0.f;
  float cc = sigmoidf_(f_) * c0 + sigmoidf_(i_) * tanhf(g_);
  float hh = sigmoidf_(o_) * tanhf(cc);
  int b = mrow >> nshift, j = mrow & ((1 << nshift) - 1);
  long nidx = (((long)b * NODES + off + j) << 8) + e;
  h[nidx] = hh;
  c[nidx] = cc;
}

// ---------------------------------------------------------------------------
// Final head: out[b] = sigmoid( t2[b,:] . w + bias )
// ---------------------------------------------------------------------------
__global__ void head_final_kernel(const float* __restrict__ t2,
                                  const float* __restrict__ w,
                                  const float* __restrict__ bsc,
                                  float* __restrict__ out, int Hin)
{
  int i = blockIdx.x * blockDim.x + threadIdx.x;
  if (i >= BATCH) return;
  float s = bsc[0];
  for (int k = 0; k < Hin; ++k) s += t2[(long)i * Hin + k] * w[k];
  out[i] = sigmoidf_(s);
}

// ---------------------------------------------------------------------------
static inline void launch_gemm(hipStream_t s,
                               const float* A, int lda, int ashift, int aoff,
                               const float* B, int ldb, int transB,
                               float* C, int ldc,
                               const float* bias, const float* rowmask,
                               int M, int N, int K, int accum, int act)
{
  dim3 g((N + 63) / 64, (M + 63) / 64);
  gemm_bf16_kernel<<<g, 128, 0, s>>>(A, lda, ashift, aoff, B, ldb, transB,
                                     C, ldc, bias, rowmask, M, N, K, accum, act);
}

extern "C" void kernel_launch(void* const* d_in, const int* in_sizes, int n_in,
                              void* d_out, int out_size, void* d_ws, size_t ws_size,
                              hipStream_t stream)
{
  (void)in_sizes; (void)n_in; (void)out_size; (void)ws_size;
  const float* op       = (const float*)d_in[0];
  const float* feat     = (const float*)d_in[1];
  const float* cond1    = (const float*)d_in[2];
  const float* cond2    = (const float*)d_in[3];
  const float* bitmap   = (const float*)d_in[4];
  const float* has_cond = (const float*)d_in[5];
  const float* W_op   = (const float*)d_in[6];
  const float* b_op   = (const float*)d_in[7];
  const float* W_feat = (const float*)d_in[8];
  const float* b_feat = (const float*)d_in[9];
  const float* W_pred = (const float*)d_in[10];
  const float* b_pred = (const float*)d_in[11];
  const float* W_bm   = (const float*)d_in[12];
  const float* b_bm   = (const float*)d_in[13];
  const float* W_ih   = (const float*)d_in[14];
  const float* b_ih   = (const float*)d_in[15];
  const float* W_hh   = (const float*)d_in[16];
  const float* b_hh   = (const float*)d_in[17];
  const float* W_wh   = (const float*)d_in[18];
  const float* W_ext  = (const float*)d_in[19];
  const float* W_a    = (const float*)d_in[20];
  const float* W_ht   = (const float*)d_in[21];
  const float* b_ht   = (const float*)d_in[22];
  const float* W_c1   = (const float*)d_in[23];
  const float* b_c1   = (const float*)d_in[24];
  const float* W_c2   = (const float*)d_in[25];
  const float* b_c2   = (const float*)d_in[26];
  const float* W_c3   = (const float*)d_in[27];
  const float* b_c3   = (const float*)d_in[28];
  const float* W_d1   = (const float*)d_in[29];
  const float* b_d1   = (const float*)d_in[30];
  const float* W_d2   = (const float*)d_in[31];
  const float* b_d2   = (const float*)d_in[32];
  const float* W_d3   = (const float*)d_in[33];
  const float* b_d3   = (const float*)d_in[34];
  float* out = (float*)d_out;

  // ---- workspace carve-up (fp32 elements) ----
  float* ws = (float*)d_ws;
  size_t o = 0;
  const size_t MALL = (size_t)BATCH * NODES;        // 64512
  float* x      = ws + o; o += MALL * XDIM;          // 64512 x 640
  float* hbuf   = ws + o; o += MALL * HLX;           // h per node
  float* cbuf   = ws + o; o += MALL * HLX;           // c per node
  float* gates  = ws + o; o += (size_t)BATCH * 32 * 1024;
  float* h0buf  = ws + o; o += (size_t)BATCH * 32 * HLX;
  float* c0buf  = ws + o; o += (size_t)BATCH * 32 * HLX;
  float* extbuf = ws + o; o += (size_t)BATCH * 32 * 64;
  float* pooled = ws + o; o += (size_t)BATCH * 32 * HLX;
  float* t1buf  = ws + o; o += (size_t)BATCH * HMX;
  float* t2buf  = ws + o; o += (size_t)BATCH * HMX;

  const int Mfull = (int)MALL;

  // ---- input projections into x = [op|feat|cond1|cond2|bitmap*mask] ----
  launch_gemm(stream, op,     32,   -1, 0, W_op,   HMX, 0, x + 0,   XDIM, b_op,   nullptr,  Mfull, HMX, 32,   0, 0);
  launch_gemm(stream, feat,   64,   -1, 0, W_feat, HMX, 0, x + 128, XDIM, b_feat, nullptr,  Mfull, HMX, 64,   0, 0);
  launch_gemm(stream, cond1,  256,  -1, 0, W_pred, HMX, 0, x + 256, XDIM, b_pred, nullptr,  Mfull, HMX, 256,  0, 0);
  launch_gemm(stream, cond2,  256,  -1, 0, W_pred, HMX, 0, x + 384, XDIM, b_pred, nullptr,  Mfull, HMX, 256,  0, 0);
  launch_gemm(stream, bitmap, 1000, -1, 0, W_bm,   HMX, 0, x + 512, XDIM, b_bm,   has_cond, Mfull, HMX, 1000, 0, 0);

  // ---- tree levels ----
  static const int OFFS_H[6] = {0, 32, 48, 56, 60, 62};
  for (int l = 0; l < 6; ++l) {
    const int nshift = 5 - l;                 // n = 2^nshift
    const int n      = 1 << nshift;
    const int off    = OFFS_H[l];
    const int Mrows  = BATCH * n;
    const int Tpw    = Mrows * HLX;

    if (l > 0) {
      const int m = (2 << l) - 2;             // 2^(l+1) - 2 descendants
      c0_kernel<<<(Tpw + 255) / 256, 256, 0, stream>>>(cbuf, c0buf, nshift, OFFS_H[l - 1], Tpw);
      // ext = xl @ W_ext   (640 -> 64)
      launch_gemm(stream, x, XDIM, nshift, off, W_ext, 64, 0, extbuf, 64, nullptr, nullptr,
                  Mrows, 64, XDIM, 0, 0);
      // attention -> pooled
      attention_kernel<<<Mrows, 128, 0, stream>>>(hbuf, extbuf, W_wh, W_a, pooled, l, nshift, m);
      // h0 = tanh(pooled @ W_ht + b_ht)
      launch_gemm(stream, pooled, HLX, -1, 0, W_ht, HLX, 0, h0buf, HLX, b_ht, nullptr,
                  Mrows, HLX, HLX, 0, 1);
    }

    // gates = xl @ W_ih^T + b_ih   (+ h0 @ W_hh^T + b_hh when l > 0)
    launch_gemm(stream, x, XDIM, nshift, off, W_ih, XDIM, 1, gates, 1024, b_ih, nullptr,
                Mrows, 1024, XDIM, 0, 0);
    if (l > 0)
      launch_gemm(stream, h0buf, HLX, -1, 0, W_hh, HLX, 1, gates, 1024, b_hh, nullptr,
                  Mrows, 1024, HLX, 1, 0);

    lstm_pointwise<<<(Tpw + 255) / 256, 256, 0, stream>>>(
        gates, c0buf, (l > 0) ? 1 : 0, (l == 0) ? b_hh : nullptr,
        hbuf, cbuf, nshift, off, Tpw);
  }

  // ---- heads on root = h[b, node 62] ----
  // cost
  launch_gemm(stream, hbuf, HLX, 0, 62, W_c1, HMX, 0, t1buf, HMX, b_c1, nullptr, BATCH, HMX, HLX, 0, 2);
  launch_gemm(stream, t1buf, HMX, -1, 0, W_c2, HMX, 0, t2buf, HMX, b_c2, nullptr, BATCH, HMX, HMX, 0, 2);
  head_final_kernel<<<(BATCH + 255) / 256, 256, 0, stream>>>(t2buf, W_c3, b_c3, out, HMX);
  // card (reuse t1/t2)
  launch_gemm(stream, hbuf, HLX, 0, 62, W_d1, HMX, 0, t1buf, HMX, b_d1, nullptr, BATCH, HMX, HLX, 0, 2);
  launch_gemm(stream, t1buf, HMX, -1, 0, W_d2, HMX, 0, t2buf, HMX, b_d2, nullptr, BATCH, HMX, HMX, 0, 2);
  head_final_kernel<<<(BATCH + 255) / 256, 256, 0, stream>>>(t2buf, W_d3, b_d3, out + BATCH, HMX);
}
